// ConvVAE_89069031784841
// MI455X (gfx1250) — matrix operand order repaired and checked by reading, not compiled, not run on hardware
//
#include <hip/hip_runtime.h>
#include <hip/hip_bf16.h>

typedef __attribute__((ext_vector_type(16))) _Float16     v16h;
typedef __attribute__((ext_vector_type(8)))  float        v8f;
typedef __attribute__((ext_vector_type(4)))  unsigned int v4u;
typedef __attribute__((ext_vector_type(4)))  int          v4i;
typedef __attribute__((ext_vector_type(8)))  int          v8i;

static __host__ __device__ inline int cdiv(int a, int b) { return (a + b - 1) / b; }

// ---------------------------------------------------------------------------
// TDM 2D tile load: global [rows x width] (f16, row stride in elements) -> LDS.
// Descriptor layout per CDNA5 ISA (D# groups 0/1; groups
// 2/3 zero => 2D tensor). Tracked by TENSORcnt.
// ---------------------------------------------------------------------------
static __device__ __forceinline__ unsigned int lds_off_of(const void* p) {
  // generic pointer to LDS: low 32 bits are the LDS byte offset (aperture in MSBs)
  return (unsigned int)(unsigned long long)(uintptr_t)p;
}

static __device__ __forceinline__ void tdm_load_f16_2d(unsigned int lds_byte_off,
                                                       const void* gptr,
                                                       unsigned int width_elems,
                                                       unsigned int tile_rows,
                                                       unsigned int row_stride_elems,
                                                       unsigned int tensor_rows) {
  unsigned long long ga = (unsigned long long)(uintptr_t)gptr;
  v4u g0;
  g0[0] = 1u;                                   // count=1, user-mode, no gather
  g0[1] = lds_byte_off;                         // lds_addr [63:32]
  g0[2] = (unsigned int)(ga & 0xFFFFFFFFu);     // global_addr[31:0]
  g0[3] = (unsigned int)((ga >> 32) & 0x1FFFFFFu) | (2u << 30);  // addr[56:32] | type=2
  v8i g1;
  g1[0] = (int)(1u << 16);                      // data_size=1 (2 bytes); mask=0
  g1[1] = (int)((width_elems & 0xFFFFu) << 16); // tensor_dim0[15:0] @ bits63:48
  g1[2] = (int)(((width_elems >> 16) & 0xFFFFu) |
                ((tensor_rows & 0xFFFFu) << 16));   // dim0[31:16] | dim1[15:0]
  g1[3] = (int)(((tensor_rows >> 16) & 0xFFFFu) |
                ((width_elems & 0xFFFFu) << 16));   // dim1[31:16] | tile_dim0
  g1[4] = (int)(tile_rows & 0xFFFFu);           // tile_dim1 (tile_dim2 = 0)
  g1[5] = (int)row_stride_elems;                // tensor_dim0_stride[31:0]
  g1[6] = 0;                                    // stride[47:32] | dim1_stride lo
  g1[7] = 0;
  v4i gz = {};
#if defined(__clang_major__) && (__clang_major__ >= 23)
  v8i gz8 = {};
  __builtin_amdgcn_tensor_load_to_lds(g0, g1, gz, gz, gz8, 0);
#else
  __builtin_amdgcn_tensor_load_to_lds(g0, g1, gz, gz, 0);
#endif
}

// ---------------------------------------------------------------------------
// Direct 3x3 stride-2 pad-1 conv + ReLU (fp32, NCHW / OIHW). Latency-bound at
// batch 1; ~20-115 MFLOP per layer.
// ---------------------------------------------------------------------------
__global__ void k_conv3x3_s2_relu(const float* __restrict__ in,
                                  const float* __restrict__ w,
                                  const float* __restrict__ b,
                                  float* __restrict__ out,
                                  int Cin, int Hin, int Win,
                                  int Cout, int Hout, int Wout) {
  int idx = blockIdx.x * blockDim.x + threadIdx.x;
  int total = Cout * Hout * Wout;
  if (idx >= total) return;
  int ox = idx % Wout;
  int t  = idx / Wout;
  int oy = t % Hout;
  int co = t / Hout;
  float acc = b[co];
  for (int ci = 0; ci < Cin; ++ci) {
    for (int ky = 0; ky < 3; ++ky) {
      int iy = 2 * oy - 1 + ky;
      if ((unsigned)iy >= (unsigned)Hin) continue;
      for (int kx = 0; kx < 3; ++kx) {
        int ix = 2 * ox - 1 + kx;
        if ((unsigned)ix >= (unsigned)Win) continue;
        acc += w[((co * Cin + ci) * 3 + ky) * 3 + kx] * in[(ci * Hin + iy) * Win + ix];
      }
    }
  }
  out[idx] = fmaxf(acc, 0.f);
}

// ---------------------------------------------------------------------------
// Q/K/V projection to f16. Q,K padded to 32 columns (head dim D<=8, rest zero)
// so the score GEMM maps exactly onto wmma 16x16x32 f16.
//   Qh/Kh: [N, 32] row-major f16 ; Vh: [N, C] row-major f16
// ---------------------------------------------------------------------------
template <int C, int D>
__global__ void k_qkv_prep(const float* __restrict__ x,  // [C, N]
                           const float* __restrict__ qw, const float* __restrict__ qb,
                           const float* __restrict__ kw, const float* __restrict__ kb,
                           const float* __restrict__ vw, const float* __restrict__ vb,
                           _Float16* __restrict__ Qh, _Float16* __restrict__ Kh,
                           _Float16* __restrict__ Vh, int N) {
  int n = blockIdx.x * blockDim.x + threadIdx.x;
  if (n >= N) return;
  float xv[C];
#pragma unroll
  for (int c = 0; c < C; ++c) xv[c] = x[c * N + n];
#pragma unroll
  for (int o = 0; o < 32; ++o) {
    float qa = 0.f, ka = 0.f;
    if (o < D) {
      qa = qb[o];
      ka = kb[o];
#pragma unroll
      for (int c = 0; c < C; ++c) {
        qa += qw[o * C + c] * xv[c];
        ka += kw[o * C + c] * xv[c];
      }
    }
    Qh[n * 32 + o] = (_Float16)qa;
    Kh[n * 32 + o] = (_Float16)ka;
  }
#pragma unroll
  for (int o = 0; o < C; ++o) {
    float va = vb[o];
#pragma unroll
    for (int c = 0; c < C; ++c) va += vw[o * C + c] * xv[c];
    Vh[n * C + o] = (_Float16)va;
  }
}

// ---------------------------------------------------------------------------
// Flash attention, one wave32 per 16-query block. Never materializes the NxN
// attention matrix (~630MB for layer 1). K/V chunks are double-buffered into
// LDS by the Tensor Data Mover (TENSORcnt) so chunk i+1's HBM latency hides
// behind chunk i's softmax VALU work. Two wmma 16x16x32 per chunk for scores,
// online softmax in the C/D layout, P bounced through LDS into A layout, then
// C/16 wmma accumulations for O += P*V.
// Writes out = gamma * softmax(QK)V + x   as fp32 [C, N].
// ---------------------------------------------------------------------------
template <int C>
__global__ void __launch_bounds__(32)
k_flash_attn(const _Float16* __restrict__ Qh, const _Float16* __restrict__ Kh,
             const _Float16* __restrict__ Vh, const float* __restrict__ x,
             const float* __restrict__ gamma, float* __restrict__ out, int N) {
  constexpr int CT = C / 16;
  const int lane = threadIdx.x;
  const int l = lane & 15;
  const int hi = lane >> 4;
  const int q0 = blockIdx.x * 16;

  __shared__ __align__(64) _Float16 kbuf[2][32 * 32];  // [buf][key][dpad]
  __shared__ __align__(64) _Float16 vbuf[2][32 * C];   // [buf][key][c]
  __shared__ __align__(64) _Float16 lds_p[16 * 32];    // P tile re-layout

  // A-matrix (queries): row = l for both half-waves; contiguous k = 16*hi + s.
  v16h Aq;
#pragma unroll
  for (int v = 0; v < 8; ++v) {
    int k0 = hi * 16 + 2 * v;
    Aq[2 * v]     = Qh[(q0 + l) * 32 + k0];
    Aq[2 * v + 1] = Qh[(q0 + l) * 32 + k0 + 1];
  }

  float m[8], lsum[8], corr[8];
#pragma unroll
  for (int r = 0; r < 8; ++r) {
    m[r] = -1e30f;
    lsum[r] = 0.f;
  }
  v8f zero = {};
  v8f Oacc[CT];
#pragma unroll
  for (int t = 0; t < CT; ++t) Oacc[t] = zero;

  // Prologue: TDM-load chunk 0 into buffer 0.
  tdm_load_f16_2d(lds_off_of(&kbuf[0][0]), Kh, 32u, 32u, 32u, (unsigned)N);
  tdm_load_f16_2d(lds_off_of(&vbuf[0][0]), Vh, (unsigned)C, 32u, (unsigned)C, (unsigned)N);

  int buf = 0;
  for (int n0 = 0; n0 < N; n0 += 32) {
    // Kick off next chunk into the other buffer, then wait for this chunk.
    if (n0 + 32 < N) {
      tdm_load_f16_2d(lds_off_of(&kbuf[buf ^ 1][0]),
                      Kh + (size_t)(n0 + 32) * 32, 32u, 32u, 32u, (unsigned)(N - n0 - 32));
      tdm_load_f16_2d(lds_off_of(&vbuf[buf ^ 1][0]),
                      Vh + (size_t)(n0 + 32) * C, (unsigned)C, 32u, (unsigned)C,
                      (unsigned)(N - n0 - 32));
      __builtin_amdgcn_s_wait_tensorcnt((short)2);
    } else {
      __builtin_amdgcn_s_wait_tensorcnt((short)0);
    }
    asm volatile("" ::: "memory");

    // B-matrix (keys): lane holds column l (key n0+l / n0+16+l), contiguous 16
    // f16 of the padded head dim -> two ds_load_b128 per tile.
    const v16h* kb = (const v16h*)&kbuf[buf][0];
    v16h Bk0 = kb[2 * l + hi];
    v16h Bk1 = kb[2 * (16 + l) + hi];

    v8f S0 = __builtin_amdgcn_wmma_f32_16x16x32_f16(false, Aq, false, Bk0, (short)0, zero, false, false);
    v8f S1 = __builtin_amdgcn_wmma_f32_16x16x32_f16(false, Aq, false, Bk1, (short)0, zero, false, false);

    // Online softmax per row (rows r+8*hi live across the 16 lanes of a half)
#pragma unroll
    for (int r = 0; r < 8; ++r) {
      float mx = fmaxf(S0[r], S1[r]);
#pragma unroll
      for (int msk = 1; msk < 16; msk <<= 1) mx = fmaxf(mx, __shfl_xor(mx, msk, 32));
      float nm = fmaxf(m[r], mx);
      float p0 = __expf(S0[r] - nm);
      float p1 = __expf(S1[r] - nm);
      corr[r] = __expf(m[r] - nm);
      float rs = p0 + p1;
#pragma unroll
      for (int msk = 1; msk < 16; msk <<= 1) rs += __shfl_xor(rs, msk, 32);
      lsum[r] = lsum[r] * corr[r] + rs;
      m[r] = nm;
      lds_p[(r + hi * 8) * 32 + l]      = (_Float16)p0;
      lds_p[(r + hi * 8) * 32 + 16 + l] = (_Float16)p1;
    }
#pragma unroll
    for (int t = 0; t < CT; ++t)
#pragma unroll
      for (int r = 0; r < 8; ++r) Oacc[t][r] *= corr[r];

    __syncthreads();  // single-wave WG: lowers to waits; orders lds_p

    // Reload P in A layout, accumulate O += P * Vchunk (V gathered from LDS)
    v16h Ap;
#pragma unroll
    for (int v = 0; v < 8; ++v) {
      int k0 = hi * 16 + 2 * v;
      Ap[2 * v]     = lds_p[l * 32 + k0];
      Ap[2 * v + 1] = lds_p[l * 32 + k0 + 1];
    }
#pragma unroll
    for (int t = 0; t < CT; ++t) {
      v16h Bv;
#pragma unroll
      for (int v = 0; v < 8; ++v) {
        int kd = hi * 16 + 2 * v;
        Bv[2 * v]     = vbuf[buf][kd * C + t * 16 + l];
        Bv[2 * v + 1] = vbuf[buf][(kd + 1) * C + t * 16 + l];
      }
      Oacc[t] = __builtin_amdgcn_wmma_f32_16x16x32_f16(false, Ap, false, Bv, (short)0, Oacc[t], false, false);
    }
    __syncthreads();
    asm volatile("" ::: "memory");
    buf ^= 1;
  }

  float g = gamma[0];
#pragma unroll
  for (int t = 0; t < CT; ++t)
#pragma unroll
    for (int r = 0; r < 8; ++r) {
      int tok = q0 + r + hi * 8;
      int c = t * 16 + l;
      float val = Oacc[t][r] / lsum[r];
      out[c * N + tok] = g * val + x[c * N + tok];
    }
}

// ---------------------------------------------------------------------------
// FC: out[j] = sum_k x[k] * w[k*M + j] + b[j]   (one block per j)
// ---------------------------------------------------------------------------
__global__ void k_fc_col(const float* __restrict__ x, const float* __restrict__ w,
                         const float* __restrict__ b, float* __restrict__ out,
                         int K, int M) {
  __shared__ float red[256];
  int j = blockIdx.x;
  float acc = 0.f;
  for (int k = threadIdx.x; k < K; k += 256) {
    if (k + 256 < K) __builtin_prefetch(&w[(size_t)(k + 256) * M + j], 0, 0);
    acc += x[k] * w[(size_t)k * M + j];
  }
  red[threadIdx.x] = acc;
  __syncthreads();
  for (int s = 128; s > 0; s >>= 1) {
    if ((int)threadIdx.x < s) red[threadIdx.x] += red[threadIdx.x + s];
    __syncthreads();
  }
  if (threadIdx.x == 0) out[j] = red[0] + b[j];
}

__global__ void k_reparam(const float* __restrict__ mu, const float* __restrict__ lv,
                          const float* __restrict__ eps, float* __restrict__ z, int n) {
  int i = blockIdx.x * blockDim.x + threadIdx.x;
  if (i < n) z[i] = mu[i] + eps[i] * __expf(0.5f * lv[i]);
}

// out[j] = relu(sum_k z[k] * w[k*M + j] + b[j]) ; coalesced across j
__global__ void k_fc_dec(const float* __restrict__ z, const float* __restrict__ w,
                         const float* __restrict__ b, float* __restrict__ out,
                         int K, int M) {
  int j = blockIdx.x * blockDim.x + threadIdx.x;
  if (j >= M) return;
  float acc = b[j];
  for (int k = 0; k < K; ++k) acc += z[k] * w[(size_t)k * M + j];
  out[j] = fmaxf(acc, 0.f);
}

// ---------------------------------------------------------------------------
// ConvTranspose2d(k=3, s=2, p=1, output_padding=1), weight [Cin, Cout, 3, 3].
// out[oy] gathers in[iy] where 2*iy = oy + 1 - ky. act: 0=none 1=relu 2=sigmoid
// ---------------------------------------------------------------------------
__global__ void k_deconv3x3_s2(const float* __restrict__ in, const float* __restrict__ w,
                               const float* __restrict__ b, float* __restrict__ out,
                               int Cin, int Hin, int Win,
                               int Cout, int Hout, int Wout, int act) {
  int idx = blockIdx.x * blockDim.x + threadIdx.x;
  int total = Cout * Hout * Wout;
  if (idx >= total) return;
  int ox = idx % Wout;
  int t  = idx / Wout;
  int oy = t % Hout;
  int co = t / Hout;
  float acc = b[co];
  for (int ky = 0; ky < 3; ++ky) {
    int ty = oy + 1 - ky;
    if (ty & 1) continue;
    int iy = ty >> 1;
    if ((unsigned)iy >= (unsigned)Hin) continue;
    for (int kx = 0; kx < 3; ++kx) {
      int tx = ox + 1 - kx;
      if (tx & 1) continue;
      int ix = tx >> 1;
      if ((unsigned)ix >= (unsigned)Win) continue;
      for (int ci = 0; ci < Cin; ++ci)
        acc += w[((ci * Cout + co) * 3 + ky) * 3 + kx] * in[(ci * Hin + iy) * Win + ix];
    }
  }
  if (act == 1) acc = fmaxf(acc, 0.f);
  else if (act == 2) acc = 1.f / (1.f + __expf(-acc));
  out[idx] = acc;
}

// ---------------------------------------------------------------------------
extern "C" void kernel_launch(void* const* d_in, const int* in_sizes, int n_in,
                              void* d_out, int out_size, void* d_ws, size_t ws_size,
                              hipStream_t stream) {
  const float* x       = (const float*)d_in[0];
  const float* eps     = (const float*)d_in[1];
  const float* conv1_w = (const float*)d_in[2];
  const float* conv1_b = (const float*)d_in[3];
  const float* q1_w = (const float*)d_in[4];
  const float* q1_b = (const float*)d_in[5];
  const float* k1_w = (const float*)d_in[6];
  const float* k1_b = (const float*)d_in[7];
  const float* v1_w = (const float*)d_in[8];
  const float* v1_b = (const float*)d_in[9];
  const float* gamma1 = (const float*)d_in[10];
  const float* conv2_w = (const float*)d_in[11];
  const float* conv2_b = (const float*)d_in[12];
  const float* q2_w = (const float*)d_in[13];
  const float* q2_b = (const float*)d_in[14];
  const float* k2_w = (const float*)d_in[15];
  const float* k2_b = (const float*)d_in[16];
  const float* v2_w = (const float*)d_in[17];
  const float* v2_b = (const float*)d_in[18];
  const float* gamma2 = (const float*)d_in[19];
  const float* conv3_w = (const float*)d_in[20];
  const float* conv3_b = (const float*)d_in[21];
  const float* fc_mu_w = (const float*)d_in[22];
  const float* fc_mu_b = (const float*)d_in[23];
  const float* fc_lv_w = (const float*)d_in[24];
  const float* fc_lv_b = (const float*)d_in[25];
  const float* fc_dec_w = (const float*)d_in[26];
  const float* fc_dec_b = (const float*)d_in[27];
  const float* dc1_w = (const float*)d_in[28];
  const float* dc1_b = (const float*)d_in[29];
  const float* dc2_w = (const float*)d_in[30];
  const float* dc2_b = (const float*)d_in[31];
  const float* dc3_w = (const float*)d_in[32];
  const float* dc3_b = (const float*)d_in[33];

  const int N1 = 112 * 112;  // 12544
  const int N2 = 56 * 56;    // 3136
  const int FLAT = 128 * 28 * 28;
  const int LATENT = 256;

  float* outp = (float*)d_out;
  float* mu_out = outp + 3 * 224 * 224;  // 150528
  float* lv_out = mu_out + LATENT;       // 150784

  // bump allocator over workspace
  char* wsp = (char*)d_ws;
  auto alloc = [&](size_t bytes) -> void* {
    void* p = (void*)wsp;
    wsp += (bytes + 255) & ~(size_t)255;
    return p;
  };
  float*    h1  = (float*)alloc(sizeof(float) * 32 * N1);
  _Float16* q1h = (_Float16*)alloc(sizeof(_Float16) * (size_t)N1 * 32);
  _Float16* k1h = (_Float16*)alloc(sizeof(_Float16) * (size_t)N1 * 32);
  _Float16* v1h = (_Float16*)alloc(sizeof(_Float16) * (size_t)N1 * 32);
  float*    a1  = (float*)alloc(sizeof(float) * 32 * N1);
  float*    h2  = (float*)alloc(sizeof(float) * 64 * N2);
  _Float16* q2h = (_Float16*)alloc(sizeof(_Float16) * (size_t)N2 * 32);
  _Float16* k2h = (_Float16*)alloc(sizeof(_Float16) * (size_t)N2 * 32);
  _Float16* v2h = (_Float16*)alloc(sizeof(_Float16) * (size_t)N2 * 64);
  float*    a2  = (float*)alloc(sizeof(float) * 64 * N2);
  float*    h3  = (float*)alloc(sizeof(float) * FLAT);
  float*    zv  = (float*)alloc(sizeof(float) * LATENT);
  float*    dv  = (float*)alloc(sizeof(float) * FLAT);
  float*    d1  = (float*)alloc(sizeof(float) * 64 * N2);
  float*    d2  = (float*)alloc(sizeof(float) * 32 * N1);
  (void)ws_size; (void)n_in; (void)in_sizes; (void)out_size;

  // encoder
  k_conv3x3_s2_relu<<<cdiv(32 * N1, 256), 256, 0, stream>>>(
      x, conv1_w, conv1_b, h1, 3, 224, 224, 32, 112, 112);
  k_qkv_prep<32, 4><<<cdiv(N1, 128), 128, 0, stream>>>(
      h1, q1_w, q1_b, k1_w, k1_b, v1_w, v1_b, q1h, k1h, v1h, N1);
  k_flash_attn<32><<<N1 / 16, 32, 0, stream>>>(q1h, k1h, v1h, h1, gamma1, a1, N1);

  k_conv3x3_s2_relu<<<cdiv(64 * N2, 256), 256, 0, stream>>>(
      a1, conv2_w, conv2_b, h2, 32, 112, 112, 64, 56, 56);
  k_qkv_prep<64, 8><<<cdiv(N2, 128), 128, 0, stream>>>(
      h2, q2_w, q2_b, k2_w, k2_b, v2_w, v2_b, q2h, k2h, v2h, N2);
  k_flash_attn<64><<<N2 / 16, 32, 0, stream>>>(q2h, k2h, v2h, h2, gamma2, a2, N2);

  k_conv3x3_s2_relu<<<cdiv(FLAT, 256), 256, 0, stream>>>(
      a2, conv3_w, conv3_b, h3, 64, 56, 56, 128, 28, 28);

  // latent
  k_fc_col<<<LATENT, 256, 0, stream>>>(h3, fc_mu_w, fc_mu_b, mu_out, FLAT, LATENT);
  k_fc_col<<<LATENT, 256, 0, stream>>>(h3, fc_lv_w, fc_lv_b, lv_out, FLAT, LATENT);
  k_reparam<<<1, 256, 0, stream>>>(mu_out, lv_out, eps, zv, LATENT);
  k_fc_dec<<<cdiv(FLAT, 256), 256, 0, stream>>>(zv, fc_dec_w, fc_dec_b, dv, LATENT, FLAT);

  // decoder
  k_deconv3x3_s2<<<cdiv(64 * N2, 256), 256, 0, stream>>>(
      dv, dc1_w, dc1_b, d1, 128, 28, 28, 64, 56, 56, 1);
  k_deconv3x3_s2<<<cdiv(32 * N1, 256), 256, 0, stream>>>(
      d1, dc2_w, dc2_b, d2, 64, 56, 56, 32, 112, 112, 1);
  k_deconv3x3_s2<<<cdiv(3 * 224 * 224, 256), 256, 0, stream>>>(
      d2, dc3_w, dc3_b, outp, 32, 112, 112, 3, 224, 224, 2);
}
